// GATModel_29850022707869
// MI455X (gfx1250) — matrix operand order repaired and checked by reading, not compiled, not run on hardware
//
#include <hip/hip_runtime.h>

// ---------------------------------------------------------------------------
// GAT (2-layer, PyG GATConv semantics) for MI455X / gfx1250.
// GEMMs: LDS-free WMMA (v_wmma_f32_16x16x32_bf16), compile-time N/K so all
// addressing is immediate-offset. A fragments loaded directly from global
// with packed bf16 conversion; B (weights) pre-swizzled into fragment order.
// Edge softmax/aggregation use f32 global atomics. Graph-capture safe.
// ---------------------------------------------------------------------------

typedef __attribute__((ext_vector_type(16))) __bf16 v16bf;
typedef __attribute__((ext_vector_type(4)))  __bf16 v4bf;
typedef __attribute__((ext_vector_type(4)))  float  v4f;
typedef __attribute__((ext_vector_type(8)))  float  v8f;

#define NEG_SLOPE 0.2f
#define GAT_EPS   1e-16f

__device__ __forceinline__ unsigned ord_f32(float f) {
  unsigned u = __float_as_uint(f);
  return (u & 0x80000000u) ? ~u : (u | 0x80000000u);
}
__device__ __forceinline__ float unord_f32(unsigned u) {
  unsigned b = (u & 0x80000000u) ? (u & 0x7FFFFFFFu) : ~u;
  return __uint_as_float(b);
}

// ---------------------------------------------------------------------------
// Pack B[K,N] (f32) into WMMA B-fragment order (bf16).
// Tile (kt, nt) covers K=kt*32..+31, N=nt*16..+15. For lane L in 0..31:
//   col = nt*16 + (L&15), kbase = kt*32 + (L>>4)*8, K = kbase+j (j<8), kbase+8+j (j>=8).
// Stored contiguously: frag[((kt*ntiles + nt)*32 + L)*16 + j].
// ---------------------------------------------------------------------------
__global__ __launch_bounds__(256) void pack_b_frags(
    const float* __restrict__ B, __bf16* __restrict__ frag, int N, int K) {
  int idx = blockIdx.x * blockDim.x + threadIdx.x;         // (tile, lane)
  int ntiles = N >> 4;
  int total = (K >> 5) * ntiles * 32;
  if (idx >= total) return;
  int lane = idx & 31;
  int tile = idx >> 5;
  int kt = tile / ntiles, nt = tile - kt * ntiles;
  int col = nt * 16 + (lane & 15);
  int kbase = kt * 32 + (lane >> 4) * 8;
  __bf16* dst = frag + (size_t)idx * 16;
#pragma unroll
  for (int j = 0; j < 16; ++j)
    dst[j] = (__bf16)B[(size_t)(kbase + j + (j & 8)) * N + col];
}

// ---------------------------------------------------------------------------
// C[M,N] = A[M,K] @ B[K,N]; A f32 row-major, B pre-packed fragments (bf16),
// C f32. N, K compile-time (N%64==0, K%32==0), M runtime (guarded stores).
// Block = 128 threads (4 waves); wave computes 16 rows x 64 cols, LDS-free.
// ---------------------------------------------------------------------------
template <int N, int K>
__global__ __launch_bounds__(128) void gemm_wmma_bf16(
    const float* __restrict__ A, const __bf16* __restrict__ Bfrag,
    float* __restrict__ C, int M) {
  constexpr int NT = N >> 4;                       // n-tiles across full C
  constexpr int KT = K >> 5;                       // k-tiles
  const int lane  = threadIdx.x & 31;
  const int wave  = threadIdx.x >> 5;
  const int half  = lane >> 4;
  const int l15   = lane & 15;
  const int row16 = blockIdx.y * 64 + wave * 16;   // wave's row base
  const int bcol  = blockIdx.x * 64;               // 4 n-tiles
  const int nt0   = bcol >> 4;

  // A row for this lane; clamp (garbage rows only feed guarded C rows).
  int arow = row16 + l15;
  if (arow >= M) arow = M - 1;
  const float* Arow = A + (size_t)arow * K + half * 8;
  const __bf16* Bbase = Bfrag + ((size_t)nt0 * 32 + lane) * 16;

  v8f acc[4] = {};

#pragma unroll
  for (int kt = 0; kt < KT; ++kt) {
    const float* ap = Arow + kt * 32;
    v4f a0 = *(const v4f*)(ap);
    v4f a1 = *(const v4f*)(ap + 4);
    v4f a2 = *(const v4f*)(ap + 16);
    v4f a3 = *(const v4f*)(ap + 20);
    v4bf c0 = __builtin_convertvector(a0, v4bf);
    v4bf c1 = __builtin_convertvector(a1, v4bf);
    v4bf c2 = __builtin_convertvector(a2, v4bf);
    v4bf c3 = __builtin_convertvector(a3, v4bf);
    v16bf afrag;
#pragma unroll
    for (int j = 0; j < 4; ++j) {
      afrag[j] = c0[j]; afrag[4 + j] = c1[j];
      afrag[8 + j] = c2[j]; afrag[12 + j] = c3[j];
    }
#pragma unroll
    for (int nb = 0; nb < 4; ++nb) {
      // constant byte offsets: (kt*NT + nb) * 512 * 2
      v16bf bfrag = *(const v16bf*)(Bbase + ((size_t)kt * NT + nb) * 512);
      acc[nb] = __builtin_amdgcn_wmma_f32_16x16x32_bf16(
          false, afrag, false, bfrag, (short)0, acc[nb], false, false);
    }
  }

  // C layout per 16x16 tile: VGPR r -> M = r + 8*half, N = lane&15.
  // One per-lane base pointer; all store offsets are compile-time constants.
  float* Cp = C + (size_t)(row16 + half * 8) * N + bcol + l15;
  if (row16 + 16 <= M) {               // wave-uniform fast path: full tile
#pragma unroll
    for (int nb = 0; nb < 4; ++nb)
#pragma unroll
      for (int r = 0; r < 8; ++r)
        Cp[r * N + nb * 16] = acc[nb][r];
  } else {                             // last row-block: guard per row
#pragma unroll
    for (int nb = 0; nb < 4; ++nb)
#pragma unroll
      for (int r = 0; r < 8; ++r)
        if (row16 + half * 8 + r < M) Cp[r * N + nb * 16] = acc[nb][r];
  }
}

// ---------------------------------------------------------------------------
// a_s[i,h] = <h[i,h,:], att_s[h,:]> ; a_d likewise.  64 channels per head.
// ---------------------------------------------------------------------------
template <int HEADS>
__global__ __launch_bounds__(256) void att_scores(
    const float* __restrict__ h, const float* __restrict__ att_s,
    const float* __restrict__ att_d, float* __restrict__ a_s,
    float* __restrict__ a_d, int n) {
  int i = blockIdx.x * blockDim.x + threadIdx.x;
  if (i >= n * HEADS) return;
  int hd = i & (HEADS - 1);
  const v4f* row = (const v4f*)(h + (size_t)i * 64);
  const v4f* ws = (const v4f*)(att_s + hd * 64);
  const v4f* wd = (const v4f*)(att_d + hd * 64);
  float s = 0.f, d = 0.f;
#pragma unroll
  for (int c = 0; c < 16; ++c) {
    v4f v = row[c], vs = ws[c], vd = wd[c];
#pragma unroll
    for (int j = 0; j < 4; ++j) { s += v[j] * vs[j]; d += v[j] * vd[j]; }
  }
  a_s[i] = s;
  a_d[i] = d;
}

// Pass 1: e = leaky_relu(a_s[src]+a_d[dst]); segment max via ordered-uint atomicMax.
template <int HEADS>
__global__ __launch_bounds__(256) void edge_score_max(
    const int* __restrict__ ei, const float* __restrict__ a_s,
    const float* __restrict__ a_d, float* __restrict__ ebuf,
    unsigned* __restrict__ mord, int E, long long total) {
  long long idx = (long long)blockIdx.x * blockDim.x + threadIdx.x;
  if (idx >= total) return;
  int eidx = (int)(idx / HEADS);              // compile-time shift
  int hd = (int)(idx - (long long)eidx * HEADS);
  int s, d;
  if (eidx < E) { s = ei[eidx]; d = ei[E + eidx]; }
  else          { s = eidx - E; d = s; }      // virtual self-loop
  float e = a_s[s * HEADS + hd] + a_d[d * HEADS + hd];
  e = (e > 0.f) ? e : NEG_SLOPE * e;
  ebuf[idx] = e;
  atomicMax(&mord[d * HEADS + hd], ord_f32(e));
}

// Pass 2: w = exp(e - max); denom += w (in place in ebuf).
template <int HEADS>
__global__ __launch_bounds__(256) void edge_exp_sum(
    const int* __restrict__ ei, float* __restrict__ ebuf,
    const unsigned* __restrict__ mord, float* __restrict__ denom,
    int E, long long total) {
  long long idx = (long long)blockIdx.x * blockDim.x + threadIdx.x;
  if (idx >= total) return;
  int eidx = (int)(idx / HEADS);
  int hd = (int)(idx - (long long)eidx * HEADS);
  int d = (eidx < E) ? ei[E + eidx] : (eidx - E);
  float m = unord_f32(mord[d * HEADS + hd]);
  float w = expf(ebuf[idx] - m);
  ebuf[idx] = w;
  atomicAdd(&denom[d * HEADS + hd], w);
}

// Pass 3 (block per edge): out[dst,h,c] += h[src,h,c] * (w / (denom + eps)).
template <int HEADS, int BLK>
__global__ __launch_bounds__(BLK) void edge_aggregate(
    const int* __restrict__ ei, const float* __restrict__ ebuf,
    const float* __restrict__ denom, const float* __restrict__ h,
    float* __restrict__ out, int E) {
  constexpr int CHT = HEADS * 64;
  int eidx = blockIdx.x;
  int s, d;
  if (eidx < E) { s = ei[eidx]; d = ei[E + eidx]; }
  else          { s = eidx - E; d = s; }
#pragma unroll
  for (int ch = threadIdx.x; ch < CHT; ch += BLK) {
    int hd = ch >> 6;
    float alpha = ebuf[(size_t)eidx * HEADS + hd] /
                  (denom[d * HEADS + hd] + GAT_EPS);
    atomicAdd(&out[(size_t)d * CHT + ch], h[(size_t)s * CHT + ch] * alpha);
  }
}

// x = elu(x + bias[col])  (layer-1 epilogue), COLS power of two.
template <int COLS>
__global__ __launch_bounds__(256) void bias_elu(
    float* __restrict__ x, const float* __restrict__ b, long long total) {
  long long idx = (long long)blockIdx.x * blockDim.x + threadIdx.x;
  if (idx >= total) return;
  int c = (int)(idx & (COLS - 1));
  float v = x[idx] + b[c];
  x[idx] = (v > 0.f) ? v : (expf(v) - 1.f);
}

// Pool: pools[g,c] += out2[n,c] + b2[c]; cnt[g] += 1 (once per node).
__global__ __launch_bounds__(256) void pool_sum(
    const float* __restrict__ out2, const float* __restrict__ b2,
    const int* __restrict__ batch, float* __restrict__ pools,
    float* __restrict__ cnt, int n) {
  int idx = blockIdx.x * blockDim.x + threadIdx.x;
  if (idx >= n * 64) return;
  int node = idx >> 6, c = idx & 63;
  int g = batch[node];
  atomicAdd(&pools[g * 64 + c], out2[idx] + b2[c]);
  if (c == 0) atomicAdd(&cnt[g], 1.0f);
}

// out[g] = <pools[g]/max(cnt,1), lin_W> + lin_b
__global__ __launch_bounds__(64) void final_linear(
    const float* __restrict__ pools, const float* __restrict__ cnt,
    const float* __restrict__ linW, const float* __restrict__ linb,
    float* __restrict__ out, int G) {
  int g = blockIdx.x * blockDim.x + threadIdx.x;
  if (g >= G) return;
  float c = cnt[g];
  c = (c > 1.f) ? c : 1.f;
  float s = 0.f;
#pragma unroll 16
  for (int i = 0; i < 64; ++i) s += pools[g * 64 + i] * linW[i];
  out[g] = s / c + linb[0];
}

// ---------------------------------------------------------------------------
extern "C" void kernel_launch(void* const* d_in, const int* in_sizes, int n_in,
                              void* d_out, int out_size, void* d_ws, size_t ws_size,
                              hipStream_t stream) {
  const float* x    = (const float*)d_in[0];
  const int*   ei   = (const int*)d_in[1];     // [2, E] : row0=src, row1=dst
  const int*   bidx = (const int*)d_in[2];     // [N]
  const float* W1   = (const float*)d_in[3];   // [128, 512]
  const float* as1  = (const float*)d_in[4];   // [8, 64]
  const float* ad1  = (const float*)d_in[5];
  const float* b1   = (const float*)d_in[6];   // [512]
  const float* W2   = (const float*)d_in[7];   // [512, 64]
  const float* as2  = (const float*)d_in[8];   // [1, 64]
  const float* ad2  = (const float*)d_in[9];
  const float* b2   = (const float*)d_in[10];  // [64]
  const float* linW = (const float*)d_in[11];  // [64, 1]
  const float* linb = (const float*)d_in[12];  // [1]
  float* out = (float*)d_out;                  // [G,1] == 64 floats

  const int Nn   = in_sizes[0] / 128;          // 50000
  const int E    = in_sizes[1] / 2;            // 800000
  const int G    = out_size;                   // 64
  const int Etot = E + Nn;                     // edges + self-loops

  // ---- workspace layout (256B aligned slices) ----
  char* ws = (char*)d_ws;
  size_t off = 0;
  auto alloc = [&](size_t bytes) -> void* {
    void* p = ws + off;
    off = (off + bytes + 255) & ~(size_t)255;
    return p;
  };
  float*    h1   = (float*)alloc((size_t)Nn * 512 * 4);
  float*    out1 = (float*)alloc((size_t)Nn * 512 * 4);
  float*    ebuf = (float*)alloc((size_t)Etot * 8 * 4);
  float*    a_s1 = (float*)alloc((size_t)Nn * 8 * 4);
  float*    a_d1 = (float*)alloc((size_t)Nn * 8 * 4);
  unsigned* m1   = (unsigned*)alloc((size_t)Nn * 8 * 4);
  float*    den1 = (float*)alloc((size_t)Nn * 8 * 4);
  float*    pools = (float*)alloc((size_t)(G * 64 + G) * 4);
  float*    cnt  = pools + G * 64;
  __bf16*   W1f  = (__bf16*)alloc((size_t)(128 / 32) * (512 / 16) * 512 * 2);
  __bf16*   W2f  = (__bf16*)alloc((size_t)(512 / 32) * (64 / 16) * 512 * 2);
  // layer-2 buffers reuse h1's region (h1 dead after layer-1 aggregation)
  float*    h2   = h1;                               // [Nn, 64]
  float*    out2 = h1 + (size_t)Nn * 64;             // [Nn, 64]
  float*    a_s2 = out2 + (size_t)Nn * 64;           // [Nn]
  float*    a_d2 = a_s2 + Nn;
  unsigned* m2   = (unsigned*)(a_d2 + Nn);
  float*    den2 = (float*)(m2 + Nn);
  float*    e2   = ebuf;                             // [Etot]
  (void)ws_size; (void)n_in;

  // ---- zero init (graph-capture safe) ----
  hipMemsetAsync(out1, 0, (size_t)Nn * 512 * 4, stream);
  hipMemsetAsync(m1,   0, (size_t)Nn * 8 * 4, stream);
  hipMemsetAsync(den1, 0, (size_t)Nn * 8 * 4, stream);
  hipMemsetAsync(pools, 0, (size_t)(G * 64 + G) * 4, stream);

  // ---- pre-swizzle weights into WMMA B-fragment order ----
  pack_b_frags<<<((128 / 32) * (512 / 16) * 32 + 255) / 256, 256, 0, stream>>>(
      W1, W1f, 512, 128);
  pack_b_frags<<<((512 / 32) * (64 / 16) * 32 + 255) / 256, 256, 0, stream>>>(
      W2, W2f, 64, 512);

  const int mb = (Nn + 63) / 64;   // 782 row tiles

  // ---- layer 1 ----
  gemm_wmma_bf16<512, 128><<<dim3(512 / 64, mb), 128, 0, stream>>>(x, W1f, h1, Nn);

  att_scores<8><<<(Nn * 8 + 255) / 256, 256, 0, stream>>>(h1, as1, ad1, a_s1, a_d1, Nn);
  {
    long long tot = (long long)Etot * 8;
    int blk = (int)((tot + 255) / 256);
    edge_score_max<8><<<blk, 256, 0, stream>>>(ei, a_s1, a_d1, ebuf, m1, E, tot);
    edge_exp_sum<8><<<blk, 256, 0, stream>>>(ei, ebuf, m1, den1, E, tot);
  }
  edge_aggregate<8, 256><<<Etot, 256, 0, stream>>>(ei, ebuf, den1, h1, out1, E);
  {
    long long tot = (long long)Nn * 512;
    bias_elu<512><<<(int)((tot + 255) / 256), 256, 0, stream>>>(out1, b1, tot);
  }

  // ---- layer 2 (heads = 1) ----
  gemm_wmma_bf16<64, 512><<<dim3(64 / 64, mb), 128, 0, stream>>>(out1, W2f, h2, Nn);

  hipMemsetAsync(out2, 0, (size_t)Nn * 64 * 4, stream);
  hipMemsetAsync(m2,   0, (size_t)Nn * 4, stream);
  hipMemsetAsync(den2, 0, (size_t)Nn * 4, stream);

  att_scores<1><<<(Nn + 255) / 256, 256, 0, stream>>>(h2, as2, ad2, a_s2, a_d2, Nn);
  {
    long long tot = (long long)Etot;
    int blk = (int)((tot + 255) / 256);
    edge_score_max<1><<<blk, 256, 0, stream>>>(ei, a_s2, a_d2, e2, m2, E, tot);
    edge_exp_sum<1><<<blk, 256, 0, stream>>>(ei, e2, m2, den2, E, tot);
  }
  edge_aggregate<1, 64><<<Etot, 64, 0, stream>>>(ei, e2, den2, h2, out2, E);

  // ---- pool + final linear ----
  pool_sum<<<(Nn * 64 + 255) / 256, 256, 0, stream>>>(out2, b2, bidx, pools, cnt, Nn);
  final_linear<<<(G + 63) / 64, 64, 0, stream>>>(pools, cnt, linW, linb, out, G);
}
